// UnifiedAttention_10660108828686
// MI455X (gfx1250) — matrix operand order repaired and checked by reading, not compiled, not run on hardware
//
#include <hip/hip_runtime.h>

// ---------------- Types / constants ----------------
typedef _Float16 h8   __attribute__((ext_vector_type(8)));
typedef _Float16 v16h __attribute__((ext_vector_type(16)));
typedef float    v8f  __attribute__((ext_vector_type(8)));
typedef float    f4   __attribute__((ext_vector_type(4)));

#define BM 128
#define BN 64
#define BK 32
#define LDPAD 40   // padded LDS row stride in halves (80B -> conflict-free b128)

static constexpr int kB = 8, kN = 2048, kC = 1024;
static constexpr int kM = kB * kN;          // 16384 rows
static constexpr int kCT = kC / BN;         // 16 column tiles per C-wide matrix

// CDNA5 async global->LDS copy (ASYNCcnt path), with safe fallback.
#if __has_builtin(__builtin_amdgcn_global_load_async_to_lds_b128) && \
    __has_builtin(__builtin_amdgcn_s_wait_asynccnt)
#define USE_ASYNC_LDS 1
#else
#define USE_ASYNC_LDS 0
#endif

#if USE_ASYNC_LDS
typedef int i4v __attribute__((vector_size(16)));
typedef __attribute__((address_space(1))) void* as1_void;
typedef __attribute__((address_space(3))) void* as3_void;
typedef __attribute__((address_space(1))) i4v*  as1_i4v;
typedef __attribute__((address_space(3))) i4v*  as3_i4v;

__device__ __forceinline__ void async_copy16(const _Float16* g, _Float16* l) {
  __builtin_amdgcn_global_load_async_to_lds_b128(
      (as1_i4v)(as1_void)(void*)const_cast<_Float16*>(g),
      (as3_i4v)(as3_void)(void*)l, 0, 0);
}
__device__ __forceinline__ void async_drain() {
  __builtin_amdgcn_s_wait_asynccnt(0);
}
#else
__device__ __forceinline__ void async_drain() {}
#endif

__device__ __forceinline__ v8f wmma16(v16h a, v16h b, v8f c) {
  // D = A(16x32 f16) * B(32x16 f16) + C(16x16 f32)
  return __builtin_amdgcn_wmma_f32_16x16x32_f16(false, a, false, b, (short)0, c,
                                                false, false);
}

__device__ __forceinline__ v16h join16(h8 lo, h8 hi) {
  v16h r;
#pragma unroll
  for (int i = 0; i < 8; ++i) { r[i] = lo[i]; r[i + 8] = hi[i]; }
  return r;
}

// ---------------- 1. transpose + f32->f16 convert ----------------
// dst[c][r] = (f16)src[r][c]; src is rows x cols, dst is cols x rows
__global__ void __launch_bounds__(256)
k_transpose_f16(const float* __restrict__ src, _Float16* __restrict__ dst,
                int rows, int cols) {
  __shared__ _Float16 tile[32][33];
  int c0 = blockIdx.x * 32, r0 = blockIdx.y * 32;
  int tx = threadIdx.x & 31, ty = threadIdx.x >> 5;  // 32x8
#pragma unroll
  for (int i = ty; i < 32; i += 8)
    tile[i][tx] = (_Float16)src[(size_t)(r0 + i) * cols + (c0 + tx)];
  __syncthreads();
#pragma unroll
  for (int i = ty; i < 32; i += 8)
    dst[(size_t)(c0 + i) * rows + (r0 + tx)] = tile[tx][i];
}

// ---------------- 2. qkv GEMM (WMMA), specialized ----------------
// QK=true : compute q & k tiles, reduce 5 per-row statistics, write partials.
// QK=false: compute v tiles, store as f16 for the projection GEMM.
// Double-buffered LDS; A staged through registers (f32->f16 convert),
// B copied via async global->LDS when available.
template <bool QK>
__global__ void __launch_bounds__(256)
k_qkv_gemm(const float* __restrict__ x, const _Float16* __restrict__ qkvT,
           _Float16* __restrict__ vh, float* __restrict__ part) {
  __shared__ __align__(16) _Float16 As[2][BM * LDPAD];
  __shared__ __align__(16) _Float16 Bs0[2][BN * LDPAD];
  __shared__ __align__(16) _Float16 Bs1[2][BN * LDPAD];

  const int tid = threadIdx.x;
  const int wave = tid >> 5, lane = tid & 31;
  const int laneHi = lane >> 4, laneLo = lane & 15;
  const int m0 = blockIdx.x * BM;
  const int ct = blockIdx.y;
  const int c0 = ct * BN;

  const _Float16* Bsrc0 = qkvT + (size_t)((QK ? 0 : 2 * kC) + c0) * kC;
  const _Float16* Bsrc1 = qkvT + (size_t)(kC + c0) * kC;

  v8f acc0[4], acc1[4];
#pragma unroll
  for (int j = 0; j < 4; ++j)
#pragma unroll
    for (int e = 0; e < 8; ++e) { acc0[j][e] = 0.f; acc1[j][e] = 0.f; }

  const int brow = tid >> 2, bkc = (tid & 3) * 8;

  f4 aReg[2][2];
#if !USE_ASYNC_LDS
  h8 bReg0, bReg1;
#endif

  auto gloadA = [&](int k0) {
#pragma unroll
    for (int i = 0; i < 2; ++i) {
      int q = tid * 2 + i, row = q >> 2, kc = (q & 3) * 8;
      const float* src = x + (size_t)(m0 + row) * kC + k0 + kc;
      aReg[i][0] = *(const f4*)src;
      aReg[i][1] = *(const f4*)(src + 4);
    }
  };
  auto loadB = [&](int k0, int buf) {
#if USE_ASYNC_LDS
    async_copy16(Bsrc0 + (size_t)brow * kC + k0 + bkc, &Bs0[buf][brow * LDPAD + bkc]);
    if (QK)
      async_copy16(Bsrc1 + (size_t)brow * kC + k0 + bkc, &Bs1[buf][brow * LDPAD + bkc]);
#else
    bReg0 = *(const h8*)(Bsrc0 + (size_t)brow * kC + k0 + bkc);
    if (QK) bReg1 = *(const h8*)(Bsrc1 + (size_t)brow * kC + k0 + bkc);
#endif
  };
  auto sstore = [&](int buf) {
#pragma unroll
    for (int i = 0; i < 2; ++i) {
      int q = tid * 2 + i, row = q >> 2, kc = (q & 3) * 8;
      h8 hv;
#pragma unroll
      for (int e = 0; e < 4; ++e) {
        hv[e] = (_Float16)aReg[i][0][e];
        hv[4 + e] = (_Float16)aReg[i][1][e];
      }
      *(h8*)&As[buf][row * LDPAD + kc] = hv;
    }
#if !USE_ASYNC_LDS
    *(h8*)&Bs0[buf][brow * LDPAD + bkc] = bReg0;
    if (QK) *(h8*)&Bs1[buf][brow * LDPAD + bkc] = bReg1;
#endif
  };

  // prologue: fill buffer 0
  gloadA(0);
  loadB(0, 0);
  sstore(0);
  async_drain();
  __syncthreads();

  const int aRow = wave * 16 + laneLo;
  const int aOff0 = aRow * LDPAD + 16 * laneHi;
  const int aOff1 = aRow * LDPAD + 16 * laneHi + 8;

  for (int k0 = 0; k0 < kC; k0 += BK) {
    const int cur = (k0 >> 5) & 1, nxt = cur ^ 1;
    const bool more = (k0 + BK) < kC;
    if (more) {
      gloadA(k0 + BK);
      loadB(k0 + BK, nxt);   // async path writes LDS[nxt] directly
    }

    h8 alo = *(const h8*)&As[cur][aOff0];
    h8 ahi = *(const h8*)&As[cur][aOff1];
    v16h aF = join16(alo, ahi);
#pragma unroll
    for (int j = 0; j < 4; ++j) {
      int n = j * 16 + laneLo;
      h8 blo = *(const h8*)&Bs0[cur][n * LDPAD + 16 * laneHi];
      h8 bhi = *(const h8*)&Bs0[cur][n * LDPAD + 16 * laneHi + 8];
      acc0[j] = wmma16(aF, join16(blo, bhi), acc0[j]);
      if (QK) {
        h8 clo = *(const h8*)&Bs1[cur][n * LDPAD + 16 * laneHi];
        h8 chi = *(const h8*)&Bs1[cur][n * LDPAD + 16 * laneHi + 8];
        acc1[j] = wmma16(aF, join16(clo, chi), acc1[j]);
      }
    }

    if (more) sstore(nxt);
    async_drain();
    __syncthreads();
  }

  if (!QK) {
    // store v tile as f16
#pragma unroll
    for (int j = 0; j < 4; ++j) {
      int n = c0 + j * 16 + laneLo;
#pragma unroll
      for (int r = 0; r < 8; ++r) {
        int row = m0 + wave * 16 + laneHi * 8 + r;
        vh[(size_t)row * kC + n] = (_Float16)acc0[j][r];
      }
    }
  } else {
    // per-row partial sums over this 64-column tile: Sq, Sk, Sqq, Skk, Sqk
#pragma unroll
    for (int r = 0; r < 8; ++r) {
      float sq = 0.f, sk = 0.f, sqq = 0.f, skk = 0.f, sqk = 0.f;
#pragma unroll
      for (int j = 0; j < 4; ++j) {
        float qv = acc0[j][r], kv = acc1[j][r];
        sq += qv; sk += kv; sqq += qv * qv; skk += kv * kv; sqk += qv * kv;
      }
#pragma unroll
      for (int mask = 1; mask < 16; mask <<= 1) {
        sq  += __shfl_xor(sq,  mask, 32);
        sk  += __shfl_xor(sk,  mask, 32);
        sqq += __shfl_xor(sqq, mask, 32);
        skk += __shfl_xor(skk, mask, 32);
        sqk += __shfl_xor(sqk, mask, 32);
      }
      if (laneLo == 0) {
        int row = m0 + wave * 16 + laneHi * 8 + r;
        part[(0 * kCT + ct) * kM + row] = sq;
        part[(1 * kCT + ct) * kM + row] = sk;
        part[(2 * kCT + ct) * kM + row] = sqq;
        part[(3 * kCT + ct) * kM + row] = skk;
        part[(4 * kCT + ct) * kM + row] = sqk;
      }
    }
  }
}

// ---------------- 3. stats -> attn_base ----------------
__global__ void __launch_bounds__(256)
k_attn_base(const float* __restrict__ part, const float* __restrict__ c1p,
            const float* __restrict__ c2p, float* __restrict__ attn) {
  int m = blockIdx.x * 256 + threadIdx.x;
  float s[5];
#pragma unroll
  for (int si = 0; si < 5; ++si) {
    float acc = 0.f;
#pragma unroll
    for (int t = 0; t < kCT; ++t) acc += part[(si * kCT + t) * kM + m];
    s[si] = acc;
  }
  const float invC = 1.f / (float)kC, invCm1 = 1.f / (float)(kC - 1);
  float muq = s[0] * invC, muk = s[1] * invC;
  float sqk = (s[4] - s[0] * s[1] * invC) * invCm1;
  float sq2 = (s[2] - s[0] * s[0] * invC) * invCm1;
  float sk2 = (s[3] - s[1] * s[1] * invC) * invCm1;
  float c1 = *c1p, c2 = *c2p;
  float num = (2.f * muq * muk + c1) * (2.f * sqk + c2);
  float den = (muq * muq + muk * muk + c1) * (sq2 + sk2 + c2);
  float r = num / (den + 1e-7f);
  attn[m] = r * r;
}

// ---------------- 4. ODE step + sigmoid + softmax (one block per batch) ----
__global__ void __launch_bounds__(256)
k_ode_softmax(const float* __restrict__ attn, const float* __restrict__ c1w,
              const float* __restrict__ gn_g, const float* __restrict__ gn_b,
              const float* __restrict__ c2w, const float* __restrict__ c2b,
              float* __restrict__ weights) {
  __shared__ float sA[kN];
  __shared__ float sH[4 * kN];
  __shared__ float red[256];
  __shared__ float stats[4];   // mu[2], rstd[2]
  const int b = blockIdx.x, t = threadIdx.x;
  const float* Ain = attn + (size_t)b * kN;

  for (int i = t; i < kN; i += 256) sA[i] = Ain[i];
  __syncthreads();

  // conv1: 1 -> 4 channels, k=3, zero pad
  for (int idx = t; idx < 4 * kN; idx += 256) {
    int ch = idx >> 11, n = idx & (kN - 1);
    float l = (n > 0) ? sA[n - 1] : 0.f;
    float m = sA[n];
    float r = (n < kN - 1) ? sA[n + 1] : 0.f;
    sH[idx] = c1w[ch * 3 + 0] * l + c1w[ch * 3 + 1] * m + c1w[ch * 3 + 2] * r;
  }
  __syncthreads();

  // GroupNorm: 2 groups of 2 channels, stats over 2*2048 elems
  for (int g = 0; g < 2; ++g) {
    float lsum = 0.f, lsq = 0.f;
    for (int i = t; i < 2 * kN; i += 256) {
      float v = sH[g * 2 * kN + i];
      lsum += v; lsq += v * v;
    }
    red[t] = lsum; __syncthreads();
    for (int off = 128; off; off >>= 1) { if (t < off) red[t] += red[t + off]; __syncthreads(); }
    float tot = red[0]; __syncthreads();
    red[t] = lsq; __syncthreads();
    for (int off = 128; off; off >>= 1) { if (t < off) red[t] += red[t + off]; __syncthreads(); }
    if (t == 0) {
      float mu = tot / (float)(2 * kN);
      float var = red[0] / (float)(2 * kN) - mu * mu;
      stats[g] = mu;
      stats[2 + g] = rsqrtf(var + 1e-5f);
    }
    __syncthreads();
  }

  // normalize + affine + relu
  for (int idx = t; idx < 4 * kN; idx += 256) {
    int ch = idx >> 11, g = ch >> 1;
    float v = (sH[idx] - stats[g]) * stats[2 + g];
    v = v * gn_g[ch] + gn_b[ch];
    sH[idx] = v > 0.f ? v : 0.f;
  }
  __syncthreads();

  // conv2 (4->1) + A += dt*y (dt=1) + sigmoid
  float bias = c2b[0];
  for (int n = t; n < kN; n += 256) {
    float y = bias;
#pragma unroll
    for (int ch = 0; ch < 4; ++ch) {
      const float* hr = &sH[ch * kN];
      float l = (n > 0) ? hr[n - 1] : 0.f;
      float m = hr[n];
      float r = (n < kN - 1) ? hr[n + 1] : 0.f;
      y += c2w[ch * 3 + 0] * l + c2w[ch * 3 + 1] * m + c2w[ch * 3 + 2] * r;
    }
    float a = sA[n] + y;
    sA[n] = 1.f / (1.f + expf(-a));   // sigmoid
  }
  __syncthreads();

  // softmax over N
  float lmax = -3.4e38f;
  for (int i = t; i < kN; i += 256) lmax = fmaxf(lmax, sA[i]);
  red[t] = lmax; __syncthreads();
  for (int off = 128; off; off >>= 1) { if (t < off) red[t] = fmaxf(red[t], red[t + off]); __syncthreads(); }
  float mx = red[0]; __syncthreads();
  float lsum = 0.f;
  for (int i = t; i < kN; i += 256) lsum += expf(sA[i] - mx);
  red[t] = lsum; __syncthreads();
  for (int off = 128; off; off >>= 1) { if (t < off) red[t] += red[t + off]; __syncthreads(); }
  float inv = 1.f / red[0];
  for (int i = t; i < kN; i += 256)
    weights[(size_t)b * kN + i] = expf(sA[i] - mx) * inv;
}

// ---------------- 5. column means of x (two stages) ----------------
__global__ void __launch_bounds__(256)
k_mean1(const float* __restrict__ x, float* __restrict__ mpart) {
  int bx = blockIdx.x;           // 256 blocks
  int b = bx >> 5, rem = bx & 31;
  int nch = rem >> 2, cch = rem & 3;
  int c = cch * 256 + threadIdx.x;
  const float* xp = x + ((size_t)(b * kN + nch * 256)) * kC + c;
  float s = 0.f;
  for (int n = 0; n < 256; ++n) s += xp[(size_t)n * kC];
  mpart[((b * 8 + nch) << 10) + c] = s;
}

__global__ void __launch_bounds__(256)
k_mean2(const float* __restrict__ mpart, float* __restrict__ m) {
  int idx = blockIdx.x * 256 + threadIdx.x;   // 8192
  int b = idx >> 10, c = idx & 1023;
  float s = 0.f;
#pragma unroll
  for (int k = 0; k < 8; ++k) s += mpart[((b * 8 + k) << 10) + c];
  m[idx] = s * (1.f / (float)kN);
}

// ---------------- 6. channel MLP (one block per batch) ----------------
__global__ void __launch_bounds__(256)
k_chmlp(const float* __restrict__ m, const float* __restrict__ w1,
        const float* __restrict__ b1, const float* __restrict__ w2,
        const float* __restrict__ b2, float* __restrict__ chw) {
  __shared__ float sm[kC];
  __shared__ float sh1[256];
  const int b = blockIdx.x, t = threadIdx.x;
  for (int i = t; i < kC; i += 256) sm[i] = m[(size_t)b * kC + i];
  __syncthreads();
  float acc = b1[t];
  for (int k = 0; k < kC; ++k) acc += sm[k] * w1[(size_t)k * 256 + t];
  sh1[t] = acc > 0.f ? acc : 0.f;
  __syncthreads();
  for (int c = t; c < kC; c += 256) {
    float a = b2[c];
    for (int j = 0; j < 256; ++j) a += sh1[j] * w2[(size_t)j * kC + c];
    chw[(size_t)b * kC + c] = 1.f / (1.f + expf(-a));
  }
}

// ---------------- 7. output GEMM (WMMA): (w*v*ch) @ proj_w + proj_b -------
__global__ void __launch_bounds__(256)
k_out_gemm(const _Float16* __restrict__ vh, const _Float16* __restrict__ projT,
           const float* __restrict__ chw, const float* __restrict__ weights,
           const float* __restrict__ proj_b, float* __restrict__ out) {
  __shared__ __align__(16) _Float16 As[2][BM * LDPAD];
  __shared__ __align__(16) _Float16 Bs[2][BN * LDPAD];

  const int tid = threadIdx.x;
  const int wave = tid >> 5, lane = tid & 31;
  const int laneHi = lane >> 4, laneLo = lane & 15;
  const int m0 = blockIdx.x * BM;
  const int c0 = blockIdx.y * BN;
  const float* chRow = chw + (size_t)(m0 >> 11) * kC;  // batch = m0/2048
  const _Float16* Bsrc = projT + (size_t)c0 * kC;

  v8f acc[4];
#pragma unroll
  for (int j = 0; j < 4; ++j)
#pragma unroll
    for (int e = 0; e < 8; ++e) acc[j][e] = 0.f;

  const int brow = tid >> 2, bkc = (tid & 3) * 8;

  h8 vReg[2];
  f4 cReg[2][2];
#if !USE_ASYNC_LDS
  h8 bReg;
#endif

  auto gloadA = [&](int k0) {
#pragma unroll
    for (int i = 0; i < 2; ++i) {
      int q = tid * 2 + i, row = q >> 2, kc = (q & 3) * 8;
      vReg[i] = *(const h8*)(vh + (size_t)(m0 + row) * kC + k0 + kc);
      const float* cp = chRow + k0 + kc;
      cReg[i][0] = *(const f4*)cp;
      cReg[i][1] = *(const f4*)(cp + 4);
    }
  };
  auto loadB = [&](int k0, int buf) {
#if USE_ASYNC_LDS
    async_copy16(Bsrc + (size_t)brow * kC + k0 + bkc, &Bs[buf][brow * LDPAD + bkc]);
#else
    bReg = *(const h8*)(Bsrc + (size_t)brow * kC + k0 + bkc);
#endif
  };
  auto sstore = [&](int buf) {
#pragma unroll
    for (int i = 0; i < 2; ++i) {
      int q = tid * 2 + i, row = q >> 2, kc = (q & 3) * 8;
      h8 hv;
#pragma unroll
      for (int e = 0; e < 4; ++e) {
        hv[e]     = (_Float16)((float)vReg[i][e]     * cReg[i][0][e]);
        hv[4 + e] = (_Float16)((float)vReg[i][4 + e] * cReg[i][1][e]);
      }
      *(h8*)&As[buf][row * LDPAD + kc] = hv;
    }
#if !USE_ASYNC_LDS
    *(h8*)&Bs[buf][brow * LDPAD + bkc] = bReg;
#endif
  };

  gloadA(0);
  loadB(0, 0);
  sstore(0);
  async_drain();
  __syncthreads();

  const int aRow = wave * 16 + laneLo;
  const int aOff0 = aRow * LDPAD + 16 * laneHi;
  const int aOff1 = aRow * LDPAD + 16 * laneHi + 8;

  for (int k0 = 0; k0 < kC; k0 += BK) {
    const int cur = (k0 >> 5) & 1, nxt = cur ^ 1;
    const bool more = (k0 + BK) < kC;
    if (more) {
      gloadA(k0 + BK);
      loadB(k0 + BK, nxt);
    }

    h8 alo = *(const h8*)&As[cur][aOff0];
    h8 ahi = *(const h8*)&As[cur][aOff1];
    v16h aF = join16(alo, ahi);
#pragma unroll
    for (int j = 0; j < 4; ++j) {
      int n = j * 16 + laneLo;
      h8 blo = *(const h8*)&Bs[cur][n * LDPAD + 16 * laneHi];
      h8 bhi = *(const h8*)&Bs[cur][n * LDPAD + 16 * laneHi + 8];
      acc[j] = wmma16(aF, join16(blo, bhi), acc[j]);
    }

    if (more) sstore(nxt);
    async_drain();
    __syncthreads();
  }

#pragma unroll
  for (int j = 0; j < 4; ++j) {
    int n = c0 + j * 16 + laneLo;
    float pb = proj_b[n];
#pragma unroll
    for (int r = 0; r < 8; ++r) {
      int row = m0 + wave * 16 + laneHi * 8 + r;
      out[(size_t)row * kC + n] = acc[j][r] * weights[row] + pb;
    }
  }
}

// ---------------- launcher ----------------
extern "C" void kernel_launch(void* const* d_in, const int* in_sizes, int n_in,
                              void* d_out, int out_size, void* d_ws, size_t ws_size,
                              hipStream_t stream) {
  const float* x       = (const float*)d_in[0];
  const float* qkv_w   = (const float*)d_in[1];
  const float* c1      = (const float*)d_in[2];
  const float* c2      = (const float*)d_in[3];
  const float* conv1_w = (const float*)d_in[4];
  const float* gn_g    = (const float*)d_in[5];
  const float* gn_b    = (const float*)d_in[6];
  const float* conv2_w = (const float*)d_in[7];
  const float* conv2_b = (const float*)d_in[8];
  const float* ch_w1   = (const float*)d_in[9];
  const float* ch_b1   = (const float*)d_in[10];
  const float* ch_w2   = (const float*)d_in[11];
  const float* ch_b2   = (const float*)d_in[12];
  const float* proj_w  = (const float*)d_in[13];
  const float* proj_b  = (const float*)d_in[14];
  float* out = (float*)d_out;

  char* ws = (char*)d_ws;
  size_t off = 0;
  _Float16* qkvT  = (_Float16*)(ws + off); off += (size_t)3 * kC * kC * 2;     // 6 MB
  _Float16* projT = (_Float16*)(ws + off); off += (size_t)kC * kC * 2;         // 2 MB
  _Float16* vh    = (_Float16*)(ws + off); off += (size_t)kM * kC * 2;         // 32 MB
  float* part  = (float*)(ws + off); off += (size_t)5 * kCT * kM * 4;          // 5 MB
  float* attn  = (float*)(ws + off); off += (size_t)kM * 4;
  float* wts   = (float*)(ws + off); off += (size_t)kM * 4;
  float* mpart = (float*)(ws + off); off += (size_t)kB * 8 * kC * 4;
  float* mbuf  = (float*)(ws + off); off += (size_t)kB * kC * 4;
  float* chw   = (float*)(ws + off); off += (size_t)kB * kC * 4;

  // 1. weight transposes + f16 convert
  k_transpose_f16<<<dim3(3 * kC / 32, kC / 32), 256, 0, stream>>>(qkv_w, qkvT, kC, 3 * kC);
  k_transpose_f16<<<dim3(kC / 32, kC / 32), 256, 0, stream>>>(proj_w, projT, kC, kC);
  // 2. qkv GEMM: q&k statistics pass, then v pass (f16 store)
  k_qkv_gemm<true><<<dim3(kM / BM, kCT), 256, 0, stream>>>(x, qkvT, vh, part);
  k_qkv_gemm<false><<<dim3(kM / BM, kCT), 256, 0, stream>>>(x, qkvT, vh, part);
  // 3. SSIM-style attention base
  k_attn_base<<<kM / 256, 256, 0, stream>>>(part, c1, c2, attn);
  // 4. ODE step + sigmoid + softmax over N
  k_ode_softmax<<<kB, 256, 0, stream>>>(attn, conv1_w, gn_g, gn_b, conv2_w, conv2_b, wts);
  // 5. batch-mean of x
  k_mean1<<<256, 256, 0, stream>>>(x, mpart);
  k_mean2<<<kB * kC / 256, 256, 0, stream>>>(mpart, mbuf);
  // 6. channel MLP
  k_chmlp<<<kB, 256, 0, stream>>>(mbuf, ch_w1, ch_b1, ch_w2, ch_b2, chw);
  // 7. scaled projection GEMM
  k_out_gemm<<<dim3(kM / BM, kC / BN), 256, 0, stream>>>(vh, projT, chw, wts, proj_b, out);
}
